// qkvAttention_38010460569640
// MI455X (gfx1250) — matrix operand order repaired and checked by reading, not compiled, not run on hardware
//
#include <hip/hip_runtime.h>
#include <hip/hip_bf16.h>

// ---------------------------------------------------------------------------
// MI455X (gfx1250) fused attention block.
//   - bf16 WMMA (v_wmma_f32_16x16x32_bf16) everywhere, fp32 accumulate:
//     problem is compute-bound (137 GFLOP vs ~5us of HBM at 23.3 TB/s).
//   - GEMM: 128x128 block tile, BK=32, double-buffered LDS staged with
//     GLOBAL_LOAD_ASYNC_TO_LDS_B128 (ASYNCcnt) when the toolchain has it.
//   - Attention: flash-style with softmax-one (implicit 0 logit -> /(1+l)),
//     1/sqrt(hd) folded into the Q projection (linearity), so Q/K staging is
//     a pure async copy.
// ---------------------------------------------------------------------------

typedef __bf16 bf16_t;
typedef __attribute__((ext_vector_type(16))) __bf16 v16bf;
typedef __attribute__((ext_vector_type(8)))  __bf16 v8bf;
typedef __attribute__((ext_vector_type(4)))  __bf16 v4bf;
typedef __attribute__((ext_vector_type(8)))  float  v8f;
typedef __attribute__((ext_vector_type(4)))  float  v4f;
typedef int v4i_vs __attribute__((vector_size(16)));  // matches builtin's int4

#define D_MODEL 1024
#define SEQ     2048
#define BATCH   4
#define HEADS   16
#define HDIM    64

#if defined(__AMDGCN__) &&                                              \
    __has_builtin(__builtin_amdgcn_global_load_async_to_lds_b128) &&    \
    __has_builtin(__builtin_amdgcn_s_wait_asynccnt)
#define HAVE_ASYNC_LDS 1
#else
#define HAVE_ASYNC_LDS 0
#endif

// 16-byte global -> LDS copy; async DMA (ASYNCcnt) when available.
static __device__ __forceinline__ void copy16_g2l(const bf16_t* g, bf16_t* l) {
#if HAVE_ASYNC_LDS
  __builtin_amdgcn_global_load_async_to_lds_b128(
      (__attribute__((address_space(1))) v4i_vs*)(void*)g,
      (__attribute__((address_space(3))) v4i_vs*)(void*)l, 0, 0);
#else
  *(v8bf*)l = *(const v8bf*)g;
#endif
}
static __device__ __forceinline__ void async_wait0() {
#if HAVE_ASYNC_LDS
  __builtin_amdgcn_s_wait_asynccnt(0);
#endif
}

static __device__ __forceinline__ v8f vzero8() {
  v8f z;
#pragma unroll
  for (int i = 0; i < 8; ++i) z[i] = 0.0f;
  return z;
}

// A-fragment (16x32 bf16): lanes 0-15 hold row M=lane, K=0..15; lanes 16-31
// hold K=16..31 (same k order as frag_b).
static __device__ __forceinline__ v16bf frag_a(const bf16_t* rowp, int lane) {
  return *(const v16bf*)(rowp + (lane >> 4) * 16);
}

// B-fragment (32x16 bf16): lane L<16 = column N=L, K=0..15; lane L+16 = K=16..31.
static __device__ __forceinline__ v16bf frag_b(const bf16_t* rowp, int lane) {
  return *(const v16bf*)(rowp + (lane >> 4) * 16);
}

// ---------------------------------------------------------------------------
// fp32 -> bf16 conversion with optional scale (vectorized x4)
// ---------------------------------------------------------------------------
__global__ void f32_to_bf16_k(const float* __restrict__ s,
                              bf16_t* __restrict__ d, int n4, float scale) {
  int i = blockIdx.x * blockDim.x + threadIdx.x;
  if (i < n4) {
    v4f v = *(const v4f*)(s + (size_t)i * 4);
    v4bf o;
#pragma unroll
    for (int j = 0; j < 4; ++j) o[j] = (bf16_t)(v[j] * scale);
    *(v4bf*)(d + (size_t)i * 4) = o;
  }
}

// ---------------------------------------------------------------------------
// Y[M,N] = X[M,K] @ W[N,K]^T + bias_scale*bias[N]
// 256 threads = 8 waves (4 M x 2 N); each wave 32x64 = 2x4 WMMA tiles.
// Double-buffered LDS; next k-tile DMA issued before current tile's WMMAs.
// x @ W^T needs no transpose: B-column n along k == W row n (contiguous).
// ---------------------------------------------------------------------------
template <bool OUT_F32>
__global__ __launch_bounds__(256) void gemm_xwt(
    const bf16_t* __restrict__ X, const bf16_t* __restrict__ W,
    const float* __restrict__ bias, float bias_scale, void* __restrict__ Yv,
    int M, int N, int K) {
  __shared__ bf16_t As[2][128 * 32];  // 2 x 8 KB
  __shared__ bf16_t Bs[2][128 * 32];  // 2 x 8 KB

  const int tid  = threadIdx.x;
  const int lane = tid & 31;
  const int wave = tid >> 5;
  const int wm = wave & 3;   // wave row (M)
  const int wn = wave >> 2;  // wave col (N)
  const int m0 = blockIdx.y * 128;
  const int n0 = blockIdx.x * 128;
  const int cl = lane & 15;

  v8f acc[2][4];
#pragma unroll
  for (int mt = 0; mt < 2; ++mt)
#pragma unroll
    for (int nt = 0; nt < 4; ++nt) acc[mt][nt] = vzero8();

  auto stage = [&](int buf, int k0) {
#pragma unroll
    for (int i = 0; i < 2; ++i) {
      int c   = tid + i * 256;  // 512 16B chunks per tile
      int row = c >> 2;
      int co  = (c & 3) * 8;
      copy16_g2l(&X[(size_t)(m0 + row) * K + k0 + co], &As[buf][row * 32 + co]);
      copy16_g2l(&W[(size_t)(n0 + row) * K + k0 + co], &Bs[buf][row * 32 + co]);
    }
  };

  const int nk = K / 32;
  stage(0, 0);
  async_wait0();
  __syncthreads();

  for (int kt = 0; kt < nk; ++kt) {
    const int cur = kt & 1;
    if (kt + 1 < nk) stage(cur ^ 1, (kt + 1) * 32);  // overlap DMA with WMMA

    v16bf af[2], bfr[4];
#pragma unroll
    for (int mt = 0; mt < 2; ++mt)
      af[mt] = frag_a(&As[cur][(wm * 32 + mt * 16 + cl) * 32], lane);
#pragma unroll
    for (int nt = 0; nt < 4; ++nt)
      bfr[nt] = frag_b(&Bs[cur][(wn * 64 + nt * 16 + cl) * 32], lane);
#pragma unroll
    for (int mt = 0; mt < 2; ++mt)
#pragma unroll
      for (int nt = 0; nt < 4; ++nt)
        acc[mt][nt] = __builtin_amdgcn_wmma_f32_16x16x32_bf16(
            false, af[mt], false, bfr[nt], (short)0, acc[mt][nt], false, false);

    async_wait0();    // next tile landed (no-op on fallback path)
    __syncthreads();  // all waves done reading cur / writing cur^1
  }

  // epilogue: C layout = elem e -> row e + 8*(lane>=16), col = lane&15
  const int hl = lane >> 4;
#pragma unroll
  for (int mt = 0; mt < 2; ++mt)
#pragma unroll
    for (int nt = 0; nt < 4; ++nt) {
      int col = n0 + wn * 64 + nt * 16 + cl;
      float bb = bias[col] * bias_scale;
#pragma unroll
      for (int e = 0; e < 8; ++e) {
        int row = m0 + wm * 32 + mt * 16 + e + hl * 8;
        float v = acc[mt][nt][e] + bb;
        if (OUT_F32)
          ((float*)Yv)[(size_t)row * N + col] = v;
        else
          ((bf16_t*)Yv)[(size_t)row * N + col] = (bf16_t)v;
      }
    }
}

// ---------------------------------------------------------------------------
// Fused flash-style attention, softmax-one (denominator = 1 + sum exp(x-m)).
// One workgroup per (b, h, 128-row q-tile); 8 waves, each owns 16 q-rows.
// Qp is already scaled by 1/sqrt(hd) (folded into the Q projection).
// ---------------------------------------------------------------------------
__global__ __launch_bounds__(256) void attn_fused(
    const bf16_t* __restrict__ Qp, const bf16_t* __restrict__ Kp,
    const bf16_t* __restrict__ Vp, bf16_t* __restrict__ Op) {
  __shared__ bf16_t Qs[128 * 64];  // 16 KB
  __shared__ bf16_t Ks[64 * 64];   //  8 KB
  __shared__ bf16_t Vts[64 * 64];  //  8 KB  V^T : [d][j]
  __shared__ bf16_t Ps[128 * 64];  // 16 KB  per-wave P rows (self-contained)

  const int tid  = threadIdx.x;
  const int lane = tid & 31;
  const int wave = tid >> 5;
  const int idx = blockIdx.x;
  const int qt = idx & 15;
  const int h  = (idx >> 4) & 15;
  const int b  = idx >> 8;
  const size_t rowBase = (size_t)b * SEQ;
  const int colBase = h * HDIM;
  const int cl = lane & 15;
  const int hl = lane >> 4;

  // stage Q tile (128 x 64) — pure copy, async-LDS eligible
#pragma unroll
  for (int i = 0; i < 4; ++i) {
    int c  = tid + i * 256;  // 1024 chunks of 8 elems
    int r  = c >> 3;
    int d0 = (c & 7) * 8;
    copy16_g2l(&Qp[(rowBase + qt * 128 + r) * D_MODEL + colBase + d0],
               &Qs[r * 64 + d0]);
  }
  async_wait0();
  __syncthreads();

  const int r0 = wave * 16;
  float m_run[8], l_run[8];
  v8f o_acc[4];
#pragma unroll
  for (int e = 0; e < 8; ++e) { m_run[e] = -1e30f; l_run[e] = 0.0f; }
#pragma unroll
  for (int dt = 0; dt < 4; ++dt) o_acc[dt] = vzero8();

  for (int kt = 0; kt < SEQ / 64; ++kt) {
    // stage K chunk (async copy) and transposed V chunk (VALU scatter)
#pragma unroll
    for (int i = 0; i < 2; ++i) {
      int c  = tid + i * 256;  // 512 chunks
      int r  = c >> 3;         // kv row 0..63
      int d0 = (c & 7) * 8;
      copy16_g2l(&Kp[(rowBase + kt * 64 + r) * D_MODEL + colBase + d0],
                 &Ks[r * 64 + d0]);
      v8bf vv =
          *(const v8bf*)&Vp[(rowBase + kt * 64 + r) * D_MODEL + colBase + d0];
#pragma unroll
      for (int j = 0; j < 8; ++j) Vts[(d0 + j) * 64 + r] = vv[j];
    }
    async_wait0();
    __syncthreads();

    // S = Q K^T  (16 q-rows x 64 kv-cols per wave), hd=64 -> 2 k-steps
    v8f s[4];
#pragma unroll
    for (int nt = 0; nt < 4; ++nt) s[nt] = vzero8();
#pragma unroll
    for (int ks = 0; ks < 2; ++ks) {
      v16bf a = frag_a(&Qs[(r0 + cl) * 64 + ks * 32], lane);
#pragma unroll
      for (int nt = 0; nt < 4; ++nt) {
        v16bf bb = frag_b(&Ks[(nt * 16 + cl) * 64 + ks * 32], lane);
        s[nt] = __builtin_amdgcn_wmma_f32_16x16x32_bf16(
            false, a, false, bb, (short)0, s[nt], false, false);
      }
    }

    // online softmax-one update; emit P (bf16) into LDS in A-layout order
#pragma unroll
    for (int e = 0; e < 8; ++e) {
      float mx = fmaxf(fmaxf(s[0][e], s[1][e]), fmaxf(s[2][e], s[3][e]));
#pragma unroll
      for (int off = 1; off < 16; off <<= 1)
        mx = fmaxf(mx, __shfl_xor(mx, off, 32));
      float mn   = fmaxf(m_run[e], mx);
      float corr = __expf(m_run[e] - mn);
      float ps   = 0.0f;
      int prow = (r0 + e + hl * 8) * 64 + cl;
#pragma unroll
      for (int nt = 0; nt < 4; ++nt) {
        float p = __expf(s[nt][e] - mn);
        ps += p;
        Ps[prow + nt * 16] = (bf16_t)p;
      }
#pragma unroll
      for (int off = 1; off < 16; off <<= 1) ps += __shfl_xor(ps, off, 32);
      l_run[e] = l_run[e] * corr + ps;
      m_run[e] = mn;
#pragma unroll
      for (int dt = 0; dt < 4; ++dt) o_acc[dt][e] *= corr;
    }

    // O += P V   (k over 64 kv rows -> 2 k-steps, 4 d-tiles)
#pragma unroll
    for (int ks = 0; ks < 2; ++ks) {
      v16bf a = frag_a(&Ps[(r0 + cl) * 64 + ks * 32], lane);
#pragma unroll
      for (int dt = 0; dt < 4; ++dt) {
        v16bf bb = frag_b(&Vts[(dt * 16 + cl) * 64 + ks * 32], lane);
        o_acc[dt] = __builtin_amdgcn_wmma_f32_16x16x32_bf16(
            false, a, false, bb, (short)0, o_acc[dt], false, false);
      }
    }
    __syncthreads();  // protect Ks/Vts for next chunk
  }

  // epilogue: divide by (1 + l) [softmax-one], store bf16 [B*SEQ, D]
#pragma unroll
  for (int e = 0; e < 8; ++e) {
    float inv = 1.0f / (1.0f + l_run[e]);
    size_t row = rowBase + qt * 128 + r0 + e + hl * 8;
#pragma unroll
    for (int dt = 0; dt < 4; ++dt)
      Op[row * D_MODEL + colBase + dt * 16 + cl] =
          (bf16_t)(o_acc[dt][e] * inv);
  }
}

// ---------------------------------------------------------------------------
extern "C" void kernel_launch(void* const* d_in, const int* in_sizes, int n_in,
                              void* d_out, int out_size, void* d_ws,
                              size_t ws_size, hipStream_t stream) {
  const float* q  = (const float*)d_in[0];
  const float* kv = (const float*)d_in[1];
  const float* Wq = (const float*)d_in[2];
  const float* bq = (const float*)d_in[3];
  const float* Wk = (const float*)d_in[4];
  const float* bk = (const float*)d_in[5];
  const float* Wv = (const float*)d_in[6];
  const float* bv = (const float*)d_in[7];
  const float* Wp = (const float*)d_in[8];
  const float* bp = (const float*)d_in[9];
  float* out = (float*)d_out;

  const size_t M = (size_t)BATCH * SEQ;  // 8192
  const float kScale = 0.125f;           // 1/sqrt(HDIM)
  char* ws = (char*)d_ws;
  size_t off = 0;
  auto alloc = [&](size_t bytes) -> void* {
    void* p = ws + off;
    off += (bytes + 255) & ~(size_t)255;
    return p;
  };
  bf16_t* q_b  = (bf16_t*)alloc(M * D_MODEL * 2);  // pre-scaled by 1/8
  bf16_t* kv_b = (bf16_t*)alloc(M * D_MODEL * 2);
  bf16_t* Wq_b = (bf16_t*)alloc((size_t)D_MODEL * D_MODEL * 2);
  bf16_t* Wk_b = (bf16_t*)alloc((size_t)D_MODEL * D_MODEL * 2);
  bf16_t* Wv_b = (bf16_t*)alloc((size_t)D_MODEL * D_MODEL * 2);
  bf16_t* Wp_b = (bf16_t*)alloc((size_t)D_MODEL * D_MODEL * 2);
  bf16_t* Qp   = (bf16_t*)alloc(M * D_MODEL * 2);
  bf16_t* Kp   = (bf16_t*)alloc(M * D_MODEL * 2);
  bf16_t* Vp   = (bf16_t*)alloc(M * D_MODEL * 2);
  bf16_t* Ao   = (bf16_t*)alloc(M * D_MODEL * 2);
  (void)ws_size; (void)in_sizes; (void)n_in; (void)out_size;

  auto cvt = [&](const float* s, bf16_t* d, size_t n, float scale) {
    int n4 = (int)(n / 4);
    f32_to_bf16_k<<<dim3((n4 + 255) / 256), dim3(256), 0, stream>>>(s, d, n4,
                                                                    scale);
  };
  cvt(q, q_b, M * D_MODEL, kScale);  // fold 1/sqrt(hd) into Q (linear)
  cvt(kv, kv_b, M * D_MODEL, 1.0f);
  cvt(Wq, Wq_b, (size_t)D_MODEL * D_MODEL, 1.0f);
  cvt(Wk, Wk_b, (size_t)D_MODEL * D_MODEL, 1.0f);
  cvt(Wv, Wv_b, (size_t)D_MODEL * D_MODEL, 1.0f);
  cvt(Wp, Wp_b, (size_t)D_MODEL * D_MODEL, 1.0f);

  dim3 gblk(256);
  dim3 ggrid(D_MODEL / 128, (unsigned)(M / 128));
  gemm_xwt<false><<<ggrid, gblk, 0, stream>>>(q_b, Wq_b, bq, kScale, Qp,
                                              (int)M, D_MODEL, D_MODEL);
  gemm_xwt<false><<<ggrid, gblk, 0, stream>>>(kv_b, Wk_b, bk, 1.0f, Kp, (int)M,
                                              D_MODEL, D_MODEL);
  gemm_xwt<false><<<ggrid, gblk, 0, stream>>>(kv_b, Wv_b, bv, 1.0f, Vp, (int)M,
                                              D_MODEL, D_MODEL);

  attn_fused<<<dim3(BATCH * HEADS * (SEQ / 128)), gblk, 0, stream>>>(Qp, Kp,
                                                                     Vp, Ao);

  gemm_xwt<true><<<ggrid, gblk, 0, stream>>>(Ao, Wp_b, bp, 1.0f, out, (int)M,
                                             D_MODEL, D_MODEL);
}
